// BasicBlock_XNOR_51092930953593
// MI455X (gfx1250) — 8-bit matrix operand order repaired and checked by reading, compiled, not run on hardware
//
#include <hip/hip_runtime.h>
#include <stdint.h>

typedef __attribute__((ext_vector_type(16))) int   v16i;
typedef __attribute__((ext_vector_type(8)))  float v8f;

#define EPSV 1e-5f
#define Bn    32
#define Cc    256
#define Hh    56
#define Ww    56
#define HP    58
#define WP    58
#define HW    (Hh*Ww)
#define PIX   (Bn*HW)
#define KDIM  2304
#define NPART 98

#define FP8_P1 0x38u
#define FP8_M1 0xB8u

__global__ __launch_bounds__(256) void k_prep_w(const float* __restrict__ w,
                                                uint8_t* __restrict__ wb,
                                                float* __restrict__ sf) {
  __shared__ float red[256];
  const int co = blockIdx.x;
  const int t  = threadIdx.x;
  const float* wc = w + (size_t)co * KDIM;
  float s = 0.f;
  for (int i = t; i < KDIM; i += 256) s += fabsf(wc[i]);
  red[t] = s; __syncthreads();
  for (int off = 128; off > 0; off >>= 1) {
    if (t < off) red[t] += red[t + off];
    __syncthreads();
  }
  if (t == 0) sf[co] = red[0] / (float)KDIM;
  for (int k = t; k < KDIM; k += 256) {
    int ci = k & 255;
    int rs = k >> 8;
    int kh = rs / 3, kw = rs - kh * 3;
    float v = wc[ci * 9 + kh * 3 + kw];
    wb[(size_t)co * KDIM + k] = (v > 0.f) ? FP8_P1 : (v < 0.f ? FP8_M1 : 0u);
  }
}

__global__ __launch_bounds__(256) void k_red1(const float* __restrict__ src,
                                              float* __restrict__ psum,
                                              float* __restrict__ psq, int nhwc) {
  __shared__ float rs_[256], rq_[256];
  const int p = blockIdx.x, c = blockIdx.y, t = threadIdx.x;
  float s = 0.f, q = 0.f;
  for (int e = 0; e < 4; ++e) {
    int j = p * 1024 + e * 256 + t;
    size_t addr;
    if (nhwc) addr = (size_t)j * Cc + c;
    else {
      int n = j / HW; int i = j - n * HW;
      addr = ((size_t)n * Cc + c) * HW + i;
    }
    float v = src[addr];
    s += v; q += v * v;
  }
  rs_[t] = s; rq_[t] = q; __syncthreads();
  for (int off = 128; off > 0; off >>= 1) {
    if (t < off) { rs_[t] += rs_[t + off]; rq_[t] += rq_[t + off]; }
    __syncthreads();
  }
  if (t == 0) { psum[c * NPART + p] = rs_[0]; psq[c * NPART + p] = rq_[0]; }
}

__global__ __launch_bounds__(128) void k_red2(const float* __restrict__ psum,
                                              const float* __restrict__ psq,
                                              float* __restrict__ mean,
                                              float* __restrict__ invstd) {
  __shared__ float rs_[128], rq_[128];
  const int c = blockIdx.x, t = threadIdx.x;
  rs_[t] = (t < NPART) ? psum[c * NPART + t] : 0.f;
  rq_[t] = (t < NPART) ? psq [c * NPART + t] : 0.f;
  __syncthreads();
  for (int off = 64; off > 0; off >>= 1) {
    if (t < off) { rs_[t] += rs_[t + off]; rq_[t] += rq_[t + off]; }
    __syncthreads();
  }
  if (t == 0) {
    float m = rs_[0] / (float)PIX;
    float v = rq_[0] / (float)PIX - m * m;
    mean[c] = m; invstd[c] = rsqrtf(v + EPSV);
  }
}

__global__ __launch_bounds__(256) void k_binarize(const float* __restrict__ src,
                                                  const float* __restrict__ mean,
                                                  const float* __restrict__ invstd,
                                                  const float* __restrict__ gamma,
                                                  const float* __restrict__ beta,
                                                  uint8_t* __restrict__ xb, int nhwc) {
  const int c  = threadIdx.x;
  int blk = blockIdx.x;
  int n  = blk / (HP * WP);
  int r2 = blk - n * (HP * WP);
  int hp = r2 / WP;
  int wp = r2 - hp * WP;
  size_t o = (size_t)blk * Cc + c;
  if (hp == 0 || hp == HP - 1 || wp == 0 || wp == WP - 1) { xb[o] = 0u; return; }
  int h = hp - 1, w = wp - 1;
  float v;
  if (nhwc) v = src[(((size_t)n * Hh + h) * Ww + w) * Cc + c];
  else      v = src[(((size_t)n * Cc + c) * Hh + h) * Ww + w];
  v = (v - mean[c]) * invstd[c] * gamma[c] + beta[c];
  xb[o] = (v > 0.f) ? FP8_P1 : (v < 0.f ? FP8_M1 : 0u);
}

__global__ __launch_bounds__(256) void k_binconv(const uint8_t* __restrict__ xb,
                                                 const uint8_t* __restrict__ wb,
                                                 const float* __restrict__ sf,
                                                 const float* __restrict__ alpha,
                                                 float* __restrict__ y) {
  const int lane = threadIdx.x & 31;
  const int wave = threadIdx.x >> 5;
  const int mw = wave >> 2;
  const int nw = wave & 3;
  const int lm = lane & 15;
  const bool hi = lane >= 16;

  const uint8_t* aptr[4];
#pragma unroll
  for (int mi = 0; mi < 4; ++mi) {
    int m = blockIdx.x * 128 + mw * 64 + mi * 16 + lm;
    int n = m / HW; int rem = m - n * HW;
    int h = rem / Ww, w = rem - (rem / Ww) * Ww;
    aptr[mi] = xb + (((size_t)(n * HP + h) * WP) + w) * Cc;
  }
  const uint8_t* bptr[2];
#pragma unroll
  for (int ni = 0; ni < 2; ++ni) {
    int co = blockIdx.y * 128 + nw * 32 + ni * 16 + lm;
    bptr[ni] = wb + (size_t)co * KDIM;
  }

  v8f acc[4][2];
#pragma unroll
  for (int mi = 0; mi < 4; ++mi)
#pragma unroll
    for (int ni = 0; ni < 2; ++ni) acc[mi][ni] = (v8f)0.f;

  const int ako = hi ? 16 : 0;
  const int bko = hi ? 16 : 0;

  for (int step = 0; step < 18; ++step) {
    const int k0  = step * 128;
    const int rs  = k0 >> 8;
    const int ci0 = k0 & 255;
    const int r = rs / 3, s = rs - (rs / 3) * 3;
    const int aoff = (r * WP + s) * Cc + ci0;

    if (step < 17) {
      const int nk0 = k0 + 128;
      const int nrs = nk0 >> 8;
      __builtin_prefetch(aptr[0] + ((nrs / 3) * WP + (nrs - (nrs / 3) * 3)) * Cc + (nk0 & 255), 0, 1);
    }

    v16i A[4];
#pragma unroll
    for (int mi = 0; mi < 4; ++mi) {
#pragma unroll
      for (int h2 = 0; h2 < 2; ++h2) {
#pragma unroll
        for (int j = 0; j < 4; ++j) {
          const int2 t = *(const int2*)(aptr[mi] + aoff + h2 * 64 + (j >> 1) * 32 + (j & 1) * 8 + ako);
          A[mi][h2 * 8 + j * 2 + 0] = t.x;
          A[mi][h2 * 8 + j * 2 + 1] = t.y;
        }
      }
    }
    v16i Bf[2];
#pragma unroll
    for (int ni = 0; ni < 2; ++ni) {
#pragma unroll
      for (int g = 0; g < 4; ++g) {
        const int4 t = *(const int4*)(bptr[ni] + k0 + g * 32 + bko);
        Bf[ni][g * 4 + 0] = t.x; Bf[ni][g * 4 + 1] = t.y;
        Bf[ni][g * 4 + 2] = t.z; Bf[ni][g * 4 + 3] = t.w;
      }
    }
#pragma unroll
    for (int mi = 0; mi < 4; ++mi)
#pragma unroll
      for (int ni = 0; ni < 2; ++ni)
        acc[mi][ni] = __builtin_amdgcn_wmma_f32_16x16x128_fp8_fp8(
            A[mi], Bf[ni], (short)0, acc[mi][ni], false, false);
  }

#pragma unroll
  for (int mi = 0; mi < 4; ++mi) {
    const int mrow = blockIdx.x * 128 + mw * 64 + mi * 16 + (hi ? 8 : 0);
#pragma unroll
    for (int ni = 0; ni < 2; ++ni) {
      const int co = blockIdx.y * 128 + nw * 32 + ni * 16 + lm;
      const float sc = sf[co], al = alpha[co];
#pragma unroll
      for (int rr = 0; rr < 8; ++rr) {
        float v = acc[mi][ni][rr] * sc;
        v = (v >= 0.f) ? v : al * v;
        y[(size_t)(mrow + rr) * Cc + co] = v;
      }
    }
  }
}

__global__ __launch_bounds__(256) void k_final(const float* __restrict__ y,
                                               const float* __restrict__ x,
                                               const float* __restrict__ mean,
                                               const float* __restrict__ invstd,
                                               const float* __restrict__ gamma,
                                               const float* __restrict__ beta,
                                               const float* __restrict__ alpha,
                                               float* __restrict__ out) {
  const int c = threadIdx.x;
  const int p = blockIdx.x;
  int n = p / HW; int i = p - n * HW;
  float v = y[(size_t)p * Cc + c];
  v = (v - mean[c]) * invstd[c] * gamma[c] + beta[c];
  size_t xo = ((size_t)n * Cc + c) * HW + i;
  v += x[xo];
  v = (v >= 0.f) ? v : alpha[c] * v;
  out[xo] = v;
}

extern "C" void kernel_launch(void* const* d_in, const int* in_sizes, int n_in,
                              void* d_out, int out_size, void* d_ws, size_t ws_size,
                              hipStream_t stream) {
  const float* x  = (const float*)d_in[0];
  const float* w1 = (const float*)d_in[1];
  const float* w2 = (const float*)d_in[2];
  const float* g1 = (const float*)d_in[3];
  const float* b1 = (const float*)d_in[4];
  const float* g2 = (const float*)d_in[5];
  const float* b2 = (const float*)d_in[6];
  const float* g3 = (const float*)d_in[7];
  const float* b3 = (const float*)d_in[8];
  const float* a1 = (const float*)d_in[9];
  const float* a2 = (const float*)d_in[10];
  const float* a3 = (const float*)d_in[11];
  float* out = (float*)d_out;

  uint8_t* ws = (uint8_t*)d_ws;
  size_t off = 0;
  auto take = [&](size_t bytes) -> uint8_t* {
    uint8_t* p = ws + off;
    off = (off + bytes + 255) & ~(size_t)255;
    return p;
  };
  float*   y      = (float*)  take((size_t)PIX * Cc * 4);
  uint8_t* xb     =           take((size_t)Bn * HP * WP * Cc);
  uint8_t* wb     =           take((size_t)Cc * KDIM);
  float*   sf     = (float*)  take(Cc * 4);
  float*   psum   = (float*)  take((size_t)Cc * NPART * 4);
  float*   psq    = (float*)  take((size_t)Cc * NPART * 4);
  float*   mean   = (float*)  take(Cc * 4);
  float*   invstd = (float*)  take(Cc * 4);
  (void)ws_size; (void)in_sizes; (void)n_in; (void)out_size;

  const dim3 redg(NPART, Cc);
  const dim3 convg(PIX / 128, Cc / 128);
  const int  bing = Bn * HP * WP;

  k_prep_w  <<<Cc, 256, 0, stream>>>(w1, wb, sf);
  k_red1    <<<redg, 256, 0, stream>>>(x, psum, psq,  0);
  k_red2    <<<Cc, 128, 0, stream>>>(psum, psq, mean, invstd);
  k_binarize<<<bing, 256, 0, stream>>>(x, mean, invstd, g1, b1, xb,  0);
  k_binconv <<<convg, 256, 0, stream>>>(xb, wb, sf, a1, y);

  k_red1    <<<redg, 256, 0, stream>>>(y, psum, psq,  1);
  k_red2    <<<Cc, 128, 0, stream>>>(psum, psq, mean, invstd);
  k_binarize<<<bing, 256, 0, stream>>>(y, mean, invstd, g2, b2, xb,  1);
  k_prep_w  <<<Cc, 256, 0, stream>>>(w2, wb, sf);
  k_binconv <<<convg, 256, 0, stream>>>(xb, wb, sf, a2, y);

  k_red1    <<<redg, 256, 0, stream>>>(y, psum, psq,  1);
  k_red2    <<<Cc, 128, 0, stream>>>(psum, psq, mean, invstd);
  k_final   <<<PIX, 256, 0, stream>>>(y, x, mean, invstd, g3, b3, a3, out);
}
